// SimpleRelativeStructureAttention_60198261621498
// MI455X (gfx1250) — matrix operand order repaired and checked by reading, not compiled, not run on hardware
//
#include <hip/hip_runtime.h>
#include <hip/hip_bf16.h>

typedef __attribute__((ext_vector_type(16))) _Float16 v16h;
typedef __attribute__((ext_vector_type(8)))  float    v8f;
typedef __attribute__((ext_vector_type(4)))  unsigned int u32x4;

#define NEG_MAX (-3.402823466e+38f)

union AFrag {
    v16h  f;
    u32x4 u4[2];
};

__device__ __forceinline__ v8f wmma_f16(v16h a, v16h b, v8f c) {
    return __builtin_amdgcn_wmma_f32_16x16x32_f16(false, a, false, b, (short)0, c, false, false);
}

__device__ __forceinline__ float rmax16(float v) {
    #pragma unroll
    for (int m = 8; m >= 1; m >>= 1) v = fmaxf(v, __shfl_xor(v, m, 16));
    return v;
}
__device__ __forceinline__ float rsum16(float v) {
    #pragma unroll
    for (int m = 8; m >= 1; m >>= 1) v += __shfl_xor(v, m, 16);
    return v;
}

// CDNA5 async global->LDS copy (16B per active lane), tracked by ASYNCcnt.
__device__ __forceinline__ void async_copy_b128(unsigned lds_off, const _Float16* g) {
    asm volatile("global_load_async_to_lds_b128 %0, %1, off"
                 :: "v"(lds_off), "v"(g) : "memory");
}
__device__ __forceinline__ void wait_async0() {
    asm volatile("s_wait_asynccnt 0x0" ::: "memory");
}

// ---------------------------------------------------------------- convert
__global__ void f32_to_f16_kernel(const float* __restrict__ s, _Float16* __restrict__ d, int n) {
    int i = blockIdx.x * blockDim.x + threadIdx.x;
    if (i < n) d[i] = (_Float16)s[i];
}

// ---------------------------------------------------------------- WMMA GEMM: D = A(MxK) * B(NxK)^T
// Block = 8 waves sharing one 32-row A stripe. A tile (32x32 f16) is staged in
// LDS via async-to-LDS DMA, double buffered: DMA of k-step i+1 overlaps WMMAs
// of k-step i. Each wave computes a 32(M) x 64(N) tile: 8 WMMAs per k-step.
// mode 1: f16 store to (b,h,n,d)   [q,k]
// mode 2: f16 store to (b,h,d,n)   [v transposed]
// mode 3: f32 store row-major      [final output]
__global__ void gemm_wmma_kernel(const _Float16* __restrict__ A, const _Float16* __restrict__ B,
                                 void* __restrict__ outp, int M, int N, int K, int mode) {
    __shared__ __align__(16) _Float16 Alds[2][32][32];   // 4 KiB double-buffered A tile

    const int lane = threadIdx.x & 31;
    const int w    = threadIdx.x >> 5;                // wave in block (0..7)
    const int tiles_n = N >> 6;                       // 64-wide N tiles
    const int grpn = tiles_n >> 3;                    // N-tile groups of 8 per M stripe
    const int tm   = blockIdx.x / grpn;               // shared by whole block
    const int tn   = (blockIdx.x % grpn) * 8 + w;
    const int l15  = lane & 15;
    const int lh   = lane >> 4;
    const int ka8  = lh * 8;                          // A lane-half K offset
    const int kb16 = lh * 16;                         // unused; B uses ka8 as well

    v8f cf[2][4];
    #pragma unroll
    for (int mi = 0; mi < 2; ++mi)
        #pragma unroll
        for (int nt = 0; nt < 4; ++nt)
            cf[mi][nt] = (v8f){0.f,0.f,0.f,0.f,0.f,0.f,0.f,0.f};

    const _Float16* bb[4];
    #pragma unroll
    for (int nt = 0; nt < 4; ++nt)
        bb[nt] = B + (size_t)(tn * 64 + nt * 16 + l15) * K;

    // staging lanes: waves 0..3 == threads 0..127; thread t copies 16B of
    // row (t>>2), segment (t&3) of the 32x32 A tile.
    const int  strow = threadIdx.x >> 2;
    const int  stseg = (threadIdx.x & 3) * 8;
    const _Float16* astage = A + (size_t)(tm * 32 + strow) * K + stseg;

    // prologue: kick off first A tile
    if (w < 4)
        async_copy_b128((unsigned)(size_t)&Alds[0][strow][stseg], astage);

    for (int kb = 0; kb < K; kb += 32) {
        const int buf = (kb >> 5) & 1;
        if (w < 4) wait_async0();                 // A tile for this k-step landed
        __syncthreads();                          // visible to all 8 waves; prev reads done
        if (w < 4 && (kb + 32 < K))               // overlap next DMA with compute
            async_copy_b128((unsigned)(size_t)&Alds[buf ^ 1][strow][stseg],
                            astage + kb + 32);

        AFrag a0, a1;
        {
            const _Float16* ar0 = &Alds[buf][l15][ka8];
            const _Float16* ar1 = &Alds[buf][16 + l15][ka8];
            a0.u4[0] = *(const u32x4*)(ar0);
            a0.u4[1] = *(const u32x4*)(ar0 + 16);
            a1.u4[0] = *(const u32x4*)(ar1);
            a1.u4[1] = *(const u32x4*)(ar1 + 16);
        }
        #pragma unroll
        for (int nt = 0; nt < 4; ++nt) {
            AFrag b;
            const _Float16* bp = bb[nt] + kb + ka8;
            b.u4[0] = *(const u32x4*)(bp);
            b.u4[1] = *(const u32x4*)(bp + 16);
            __builtin_prefetch(bp + 256, 0, 3);   // WGP-scope, ~8 K-steps ahead
            cf[0][nt] = wmma_f16(a0.f, b.f, cf[0][nt]);
            cf[1][nt] = wmma_f16(a1.f, b.f, cf[1][nt]);
        }
        __syncthreads();                          // A tile fully consumed before rewrite
    }

    // C/D layout: lanes 0-15 -> N=lane, M=r ; lanes 16-31 -> N=lane-16, M=r+8
    #pragma unroll
    for (int mi = 0; mi < 2; ++mi) {
        #pragma unroll
        for (int nt = 0; nt < 4; ++nt) {
            #pragma unroll
            for (int r = 0; r < 8; ++r) {
                const int m = tm * 32 + mi * 16 + r + 8 * lh;
                const int n = tn * 64 + nt * 16 + l15;
                const float val = cf[mi][nt][r];
                if (mode == 3) {
                    ((float*)outp)[(size_t)m * N + n] = val;
                } else {
                    const int bbx = m >> 10, ii = m & 1023;   // M = b*1024 + i
                    const int hh  = n >> 6,  dd = n & 63;     // N = h*64 + d
                    _Float16* o = (_Float16*)outp;
                    if (mode == 1)
                        o[(((size_t)(bbx * 16 + hh)) * 1024 + ii) * 64 + dd] = (_Float16)val;
                    else
                        o[(((size_t)(bbx * 16 + hh)) * 64 + dd) * 1024 + ii] = (_Float16)val;
                }
            }
        }
    }
}

// ---------------------------------------------------------------- fused causal attention
// grid: b * (n/16) * 2 blocks; block: 256 threads = 8 waves; wave w handles head hg*8+w.
// Structure-matrix index tiles staged in LDS once per block (shared by all 8 heads).
__global__ void attn_wmma_kernel(const _Float16* __restrict__ qh, const _Float16* __restrict__ kh,
                                 const _Float16* __restrict__ vth,
                                 const float* __restrict__ E_idx, const float* __restrict__ E_bar,
                                 const float* __restrict__ E_pos, const float* __restrict__ E_oct,
                                 const float* __restrict__ E_semi,
                                 const int* __restrict__ barM, const int* __restrict__ posM,
                                 const int* __restrict__ octM, const int* __restrict__ semM,
                                 _Float16* __restrict__ oh) {
    __shared__ __align__(16) _Float16 Plds[8][16][32];   // per-wave P staging (8 KiB)
    __shared__ int idxT[4][16][32];                      // bar/pos/oct/semi tile (8 KiB)

    const int n = 1024, d = 64, H = 16;
    const int hg    = blockIdx.x & 1;         // head group (0: h 0-7, 1: h 8-15)
    const int itile = (blockIdx.x >> 1) & 63; // i tile (16 rows)
    const int bidx  = blockIdx.x >> 7;        // batch
    const int i0    = itile * 16;
    const int w     = threadIdx.x >> 5;       // wave in block
    const int lane  = threadIdx.x & 31;
    const int h     = hg * 8 + w;
    const int bh    = bidx * H + h;
    const int l15   = lane & 15;
    const int lh    = lane >> 4;

    // ---- preload Q A-fragments for both 32-wide K chunks of d=64
    AFrag qa[2];
    {
        const int row = i0 + l15;                          // A-frag: M = lane%16
        const _Float16* qp = qh + ((size_t)bh * n + row) * d + lh * 8;
        qa[0].u4[0] = *(const u32x4*)(qp);
        qa[0].u4[1] = *(const u32x4*)(qp + 16);
        qa[1].u4[0] = *(const u32x4*)(qp + 32);
        qa[1].u4[1] = *(const u32x4*)(qp + 48);
    }

    v8f of[4];
    #pragma unroll
    for (int t = 0; t < 4; ++t) of[t] = (v8f){0.f,0.f,0.f,0.f,0.f,0.f,0.f,0.f};
    float mrun[8], lrun[8];
    #pragma unroll
    for (int r = 0; r < 8; ++r) { mrun[r] = NEG_MAX; lrun[r] = 0.f; }

    const int jt_max = (i0 + 15) >> 5;                      // causal bound (32-wide tiles)
    for (int jt = 0; jt <= jt_max; ++jt) {
        const int jbase = jt * 32;

        // ---- cooperative staging of the 16x32 structure-index tiles (once per block)
        {
            const int p0 = threadIdx.x;                    // positions p0 and p0+256 of 512
            #pragma unroll
            for (int q = 0; q < 2; ++q) {
                const int p = p0 + q * 256;
                const int m = p >> 5, c = p & 31;
                const size_t e = ((size_t)bidx * n + (i0 + m)) * n + (jbase + c);
                idxT[0][m][c] = barM[e];
                idxT[1][m][c] = posM[e];
                idxT[2][m][c] = octM[e];
                idxT[3][m][c] = semM[e];
            }
        }
        __syncthreads();

        // ---- S = Q * K^T (16 x 32) : two 16x16 sub-tiles, K-loop over d in 2 chunks
        v8f sf[2];
        #pragma unroll
        for (int sub = 0; sub < 2; ++sub) {
            sf[sub] = (v8f){0.f,0.f,0.f,0.f,0.f,0.f,0.f,0.f};
            const int jrow = jbase + sub * 16 + l15;        // B-frag: N = lane%16
            #pragma unroll
            for (int c = 0; c < 2; ++c) {
                AFrag kb;
                const _Float16* kp = kh + ((size_t)bh * n + jrow) * d + c * 32 + lh * 8;
                kb.u4[0] = *(const u32x4*)(kp);
                kb.u4[1] = *(const u32x4*)(kp + 16);
                sf[sub] = wmma_f16(qa[c].f, kb.f, sf[sub]);
            }
        }

        // ---- bias (indices from LDS) + causal mask + online softmax
        #pragma unroll
        for (int r = 0; r < 8; ++r) {
            const int m  = r + 8 * lh;                      // C layout row
            const int ig = i0 + m;
            float s0, s1;
            {
                const int c = l15, jg = jbase + c;
                if (jg <= ig) {
                    float bias = E_idx[min(ig - jg, 1023) * H + h]
                               + E_bar [(idxT[0][m][c] + 1) * H + h]
                               + E_pos [(idxT[1][m][c] + 1) * H + h]
                               + E_oct [(idxT[2][m][c] + 1) * H + h]
                               + E_semi[(idxT[3][m][c] + 1) * H + h];
                    s0 = (sf[0][r] + bias) * 0.125f;
                } else s0 = NEG_MAX;
            }
            {
                const int c = l15 + 16, jg = jbase + c;
                if (jg <= ig) {
                    float bias = E_idx[min(ig - jg, 1023) * H + h]
                               + E_bar [(idxT[0][m][c] + 1) * H + h]
                               + E_pos [(idxT[1][m][c] + 1) * H + h]
                               + E_oct [(idxT[2][m][c] + 1) * H + h]
                               + E_semi[(idxT[3][m][c] + 1) * H + h];
                    s1 = (sf[1][r] + bias) * 0.125f;
                } else s1 = NEG_MAX;
            }
            const float tmax = rmax16(fmaxf(s0, s1));
            const float mnew = fmaxf(mrun[r], tmax);
            const float alpha = __expf(mrun[r] - mnew);
            const float p0 = __expf(s0 - mnew);
            const float p1 = __expf(s1 - mnew);
            lrun[r] = lrun[r] * alpha + rsum16(p0 + p1);
            mrun[r] = mnew;
            #pragma unroll
            for (int t = 0; t < 4; ++t) of[t][r] *= alpha;
            Plds[w][m][l15]      = (_Float16)p0;
            Plds[w][m][l15 + 16] = (_Float16)p1;
        }

        // wave-internal LDS write->read hazard across lanes
        asm volatile("s_wait_dscnt 0" ::: "memory");

        // ---- P A-fragment from LDS (C-layout -> A-layout lane permutation)
        AFrag pa;
        {
            const _Float16* pp = &Plds[w][l15][lh * 8];
            pa.u4[0] = *(const u32x4*)(pp);
            pa.u4[1] = *(const u32x4*)(pp + 16);
        }

        // ---- O += P * V  (V stored transposed: (b,h,d,n) so K(=j) is contiguous)
        #pragma unroll
        for (int t = 0; t < 4; ++t) {
            AFrag vb;
            const int dcol = t * 16 + l15;
            const _Float16* vp = vth + ((size_t)bh * d + dcol) * n + jbase + lh * 8;
            vb.u4[0] = *(const u32x4*)(vp);
            vb.u4[1] = *(const u32x4*)(vp + 16);
            of[t] = wmma_f16(pa.f, vb.f, of[t]);
        }
        __syncthreads();   // protect idxT before next iteration's staging
    }

    // ---- epilogue: normalize, store f16 at (b, i, h*64 + d) == (b*n) x 1024
    #pragma unroll
    for (int r = 0; r < 8; ++r) {
        const int m  = r + 8 * lh;
        const int ig = i0 + m;
        const float inv = 1.0f / lrun[r];
        #pragma unroll
        for (int t = 0; t < 4; ++t) {
            const int col = h * 64 + t * 16 + l15;
            oh[((size_t)bidx * n + ig) * 1024 + col] = (_Float16)(of[t][r] * inv);
        }
    }
}

// ---------------------------------------------------------------- launcher
extern "C" void kernel_launch(void* const* d_in, const int* in_sizes, int n_in,
                              void* d_out, int out_size, void* d_ws, size_t ws_size,
                              hipStream_t stream) {
    const float* x    = (const float*)d_in[0];
    const float* Wq   = (const float*)d_in[1];
    const float* Wk   = (const float*)d_in[2];
    const float* Wv   = (const float*)d_in[3];
    const float* Wo   = (const float*)d_in[4];
    const float* Eidx = (const float*)d_in[5];
    const float* Ebar = (const float*)d_in[6];
    const float* Epos = (const float*)d_in[7];
    const float* Eoct = (const float*)d_in[8];
    const float* Esem = (const float*)d_in[9];
    const int* barM   = (const int*)d_in[10];
    const int* posM   = (const int*)d_in[11];
    const int* octM   = (const int*)d_in[12];
    const int* semM   = (const int*)d_in[13];

    const int B = 2, N = 1024, DIM = 1024;
    const int M = B * N;                  // 2048
    const size_t MB = 1u << 20;
    char* w = (char*)d_ws;
    _Float16* xh  = (_Float16*)(w + 0 * MB);   // 4 MiB
    _Float16* wqh = (_Float16*)(w + 4 * MB);   // 2 MiB
    _Float16* wkh = (_Float16*)(w + 6 * MB);
    _Float16* wvh = (_Float16*)(w + 8 * MB);
    _Float16* woh = (_Float16*)(w + 10 * MB);
    _Float16* qh  = (_Float16*)(w + 12 * MB);  // 4 MiB (b,h,n,d)
    _Float16* kh  = (_Float16*)(w + 16 * MB);  // 4 MiB (b,h,n,d)
    _Float16* vth = (_Float16*)(w + 20 * MB);  // 4 MiB (b,h,d,n)
    _Float16* oh  = (_Float16*)(w + 24 * MB);  // 4 MiB (b*n, dim)

    // 1) f32 -> f16 conversions
    const int nx = M * DIM, nw = DIM * DIM;
    f32_to_f16_kernel<<<(nx + 255) / 256, 256, 0, stream>>>(x,  xh,  nx);
    f32_to_f16_kernel<<<(nw + 255) / 256, 256, 0, stream>>>(Wq, wqh, nw);
    f32_to_f16_kernel<<<(nw + 255) / 256, 256, 0, stream>>>(Wk, wkh, nw);
    f32_to_f16_kernel<<<(nw + 255) / 256, 256, 0, stream>>>(Wv, wvh, nw);
    f32_to_f16_kernel<<<(nw + 255) / 256, 256, 0, stream>>>(Wo, woh, nw);

    // 2) projections: 2048x1024x1024; 32x64 tile per wave, 8 waves/block,
    //    A stripe async-staged in LDS per block
    const int gblocks = (M / 32) * ((DIM / 64) / 8);   // 128
    gemm_wmma_kernel<<<gblocks, 256, 0, stream>>>(xh, wqh, qh,  M, DIM, DIM, 1);
    gemm_wmma_kernel<<<gblocks, 256, 0, stream>>>(xh, wkh, kh,  M, DIM, DIM, 1);
    gemm_wmma_kernel<<<gblocks, 256, 0, stream>>>(xh, wvh, vth, M, DIM, DIM, 2);

    // 3) fused attention: one block per (b, i-tile, head-group), one wave per head
    attn_wmma_kernel<<<B * (N / 16) * 2, 256, 0, stream>>>(qh, kh, vth,
                                                           Eidx, Ebar, Epos, Eoct, Esem,
                                                           barM, posM, octM, semM, oh);

    // 4) output projection -> f32 d_out
    gemm_wmma_kernel<<<gblocks, 256, 0, stream>>>(oh, woh, d_out, M, DIM, DIM, 3);
}
